// ECC_472446403147
// MI455X (gfx1250) — matrix operand order repaired and checked by reading, not compiled, not run on hardware
//
#include <hip/hip_runtime.h>

typedef __attribute__((ext_vector_type(16))) _Float16 v16h;
typedef __attribute__((ext_vector_type(8)))  _Float16 v8h;
typedef __attribute__((ext_vector_type(8)))  float    v8f;

#define N_NODES 25000
#define N_EDGES 250000
#define F_IN 32
#define F_OUT 32
#define EDGE_DIM 6
#define H1 64
#define H2 128

// ---- workspace layout (bytes) ----
static constexpr size_t XH_OFF  = 0;           // x as f16      : 25000*32*2   = 1,600,000
static constexpr size_t H2_OFF  = 1600000;     // h2 as f16     : 250000*128*2 = 64,000,000
static constexpr size_t AGG_OFF = 65600000;    // agg f32       : 800000*4     = 3,200,000
static constexpr size_t DEG_OFF = 68800000;    // deg f32       : 25000*4      = 100,000 (pad)
static constexpr size_t BD_OFF  = 68900352;    // bd  f32       : 800000*4     = 3,200,000

__device__ __forceinline__ v16h cat8(v8h lo, v8h hi) {
  v16h r;
  #pragma unroll
  for (int i = 0; i < 8; ++i) { r[i] = lo[i]; r[i + 8] = hi[i]; }
  return r;
}

// ---- prep: x->f16, bd[n][o] = dot(b3_o, x_n), zero agg/deg ----
__launch_bounds__(256)
__global__ void ecc_prep_kernel(const float* __restrict__ x, const float* __restrict__ b3,
                                _Float16* __restrict__ xh, float* __restrict__ agg,
                                float* __restrict__ deg, float* __restrict__ bd) {
  int i = blockIdx.x * blockDim.x + threadIdx.x;
  if (i < N_NODES * F_IN) {
    xh[i]  = (_Float16)x[i];
    agg[i] = 0.0f;
    int n = i / F_OUT, o = i % F_OUT;
    float s = 0.f;
    #pragma unroll
    for (int ii = 0; ii < F_IN; ++ii) s += b3[o * F_IN + ii] * x[n * F_IN + ii];
    bd[i] = s;
  }
  if (i < N_NODES) deg[i] = 0.f;
}

// ---- degree count ----
__launch_bounds__(256)
__global__ void ecc_deg_kernel(const int* __restrict__ dst, float* __restrict__ deg) {
  int e = blockIdx.x * blockDim.x + threadIdx.x;
  if (e < N_EDGES) atomicAdd(&deg[dst[e]], 1.0f);
}

// ---- layers 1+2: edge_attr -> h2 (f16, [E][128]) ----
__launch_bounds__(256)
__global__ void ecc_mlp12_kernel(const float* __restrict__ edge_attr,
                                 const float* __restrict__ W1, const float* __restrict__ b1,
                                 const float* __restrict__ W2, const float* __restrict__ b2,
                                 _Float16* __restrict__ h2g) {
  __shared__ float    w1s[H1 * EDGE_DIM];
  __shared__ float    b1s[H1];
  __shared__ _Float16 w2s[H2 * H1];
  __shared__ float    b2s[H2];
  __shared__ _Float16 h1s[8][16 * H1];   // per-wave staging
  int tid = threadIdx.x;
  for (int i = tid; i < H1 * EDGE_DIM; i += 256) w1s[i] = W1[i];
  for (int i = tid; i < H1; i += 256)            b1s[i] = b1[i];
  for (int i = tid; i < H2 * H1; i += 256)       w2s[i] = (_Float16)W2[i];
  for (int i = tid; i < H2; i += 256)            b2s[i] = b2[i];
  __syncthreads();

  int lane = tid & 31, wv = tid >> 5;
  int lq = lane & 15, half = lane >> 4;
  int nTiles = (N_EDGES + 15) / 16;
  int wglob = blockIdx.x * 8 + wv;
  int wstride = gridDim.x * 8;

  for (int t = wglob; t < nTiles; t += wstride) {
    int tb = t * 16;
    int ec = min(tb + lq, N_EDGES - 1);
    float ea[EDGE_DIM];
    #pragma unroll
    for (int d = 0; d < EDGE_DIM; ++d) ea[d] = edge_attr[ec * EDGE_DIM + d];
    // layer 1: lane computes h1[lq][half*32 + jj]
    #pragma unroll
    for (int jj = 0; jj < 32; ++jj) {
      int j = half * 32 + jj;
      float a = b1s[j];
      #pragma unroll
      for (int d = 0; d < EDGE_DIM; ++d) a += ea[d] * w1s[j * EDGE_DIM + d];
      h1s[wv][lq * H1 + j] = (_Float16)fmaxf(a, 0.f);
    }
    asm volatile("s_wait_dscnt 0" ::: "memory");  // wave-private LDS: in-order DS, just fence
    // h1 A-fragments (A layout: K = 32f + 8*half + {0..7, 16..23})
    v16h h1f[2];
    #pragma unroll
    for (int f = 0; f < 2; ++f) {
      const v8h* p = (const v8h*)&h1s[wv][lq * H1 + 32 * f + 8 * half];
      h1f[f] = cat8(p[0], p[2]);
    }
    // layer 2: C[16e,128] = h1 @ W2^T  (B layout: K = 32f + 8*half + {0..7, 16..23})
    #pragma unroll
    for (int nt = 0; nt < 8; ++nt) {
      int n = nt * 16 + lq;
      v8f acc = {};
      #pragma unroll
      for (int f = 0; f < 2; ++f) {
        const v8h* q = (const v8h*)&w2s[n * H1 + 32 * f + 8 * half];
        v16h wf = cat8(q[0], q[2]);
        acc = __builtin_amdgcn_wmma_f32_16x16x32_f16(false, h1f[f], false, wf,
                                                     (short)0, acc, false, false);
      }
      float bias = b2s[n];
      #pragma unroll
      for (int r = 0; r < 8; ++r) {
        int ee = tb + r + 8 * half;             // C layout: row = r + 8*half, col = n
        if (ee < N_EDGES) h2g[ee * H2 + n] = (_Float16)fmaxf(acc[r] + bias, 0.f);
      }
    }
  }
}

// ---- main fused kernel: theta = h2 @ W3^T (+b3), msg = theta . x[src], scatter-add ----
// blockIdx.y selects 8 output channels (o); W3 slice (256 rows x 128) f16 in LDS,
// rows PERMUTED so the f32 C-layout of theta^T packs in-lane into the f16 B-operand
// layout of the second WMMA (whose diagonal is msg[e,o]).
__launch_bounds__(256)
__global__ void ecc_main_kernel(const _Float16* __restrict__ h2g,
                                const _Float16* __restrict__ xh,
                                const float* __restrict__ W3,
                                const float* __restrict__ bd,
                                const int* __restrict__ edge_src,
                                const int* __restrict__ edge_dst,
                                float* __restrict__ agg) {
  __shared__ _Float16 w3s[256 * H2];   // 64 KB
  int tid = threadIdx.x;
  int oBase = blockIdx.y * 8;
  int rBase = oBase * F_IN;
  // stage permuted rows: lds row l=(oo,hh,m) <- global row rBase + oo*32 + hh*8 + (m<8? m : m+8)
  for (int l = tid; l < 256; l += 256) {
    int oo = l >> 5, hh = (l >> 4) & 1, m = l & 15;
    int grow = rBase + oo * 32 + hh * 8 + (m < 8 ? m : m + 8);
    const float* srcp = W3 + grow * H2;
    _Float16* dstp = &w3s[l * H2];
    for (int k = 0; k < H2; ++k) dstp[k] = (_Float16)srcp[k];
  }
  __syncthreads();

  int lane = tid & 31, wv = tid >> 5;
  int lq = lane & 15, half = lane >> 4;
  int rsel = lq & 7;
  bool diagActive = (half == 0) ? (lq < 8) : (lq >= 8);  // diag (e,e): row=r+8*half == col=lq

  int nChunks = (N_EDGES + 31) / 32;
  int wglob = blockIdx.x * 8 + wv;
  int wstride = gridDim.x * 8;

  for (int ch = wglob; ch < nChunks; ch += wstride) {
    int cb = ch * 32;
    int srcn[2], dstn[2]; bool ok[2];
    v16h h2f[2][4], xf[2];
    #pragma unroll
    for (int g = 0; g < 2; ++g) {
      int e = cb + g * 16 + lq;
      ok[g] = e < N_EDGES;
      int ec = min(e, N_EDGES - 1);
      srcn[g] = edge_src[ec];
      dstn[g] = edge_dst[ec];
      const v8h* ph = (const v8h*)(h2g + (size_t)ec * H2);  // B layout: K=32f+8*half+{0..7,16..23}
      #pragma unroll
      for (int f = 0; f < 4; ++f)
        h2f[g][f] = cat8(ph[4 * f + half], ph[4 * f + half + 2]);
      const v8h* px = (const v8h*)(xh + (size_t)srcn[g] * F_IN); // A layout
      xf[g] = cat8(px[2 * half], px[2 * half + 1]);
    }
    #pragma unroll 1
    for (int oo = 0; oo < 8; ++oo) {
      v8f acc[2][2] = {};   // [hh][g]  theta^T tiles (rows = permuted i, cols = edges)
      #pragma unroll
      for (int hh = 0; hh < 2; ++hh) {
        #pragma unroll
        for (int f = 0; f < 4; ++f) {
          const v8h* pw = (const v8h*)&w3s[(oo * 32 + hh * 16 + lq) * H2 + 32 * f + 8 * half];
          v16h wf = cat8(pw[0], pw[2]);       // A layout frag of W3 (shared by both groups)
          acc[hh][0] = __builtin_amdgcn_wmma_f32_16x16x32_f16(false, wf, false, h2f[0][f],
                                                              (short)0, acc[hh][0], false, false);
          acc[hh][1] = __builtin_amdgcn_wmma_f32_16x16x32_f16(false, wf, false, h2f[1][f],
                                                              (short)0, acc[hh][1], false, false);
        }
      }
      int o = oBase + oo;
      #pragma unroll
      for (int g = 0; g < 2; ++g) {
        // pack theta^T (f32 C-layout) -> f16 B-operand; permuted W3 rows make this in-lane:
        // element j (K = 16*half + j): j<8 from hh=0 acc, j>=8 from hh=1 acc.
        v16h b2t;
        #pragma unroll
        for (int j = 0; j < 8; ++j) {
          b2t[j]     = (_Float16)acc[0][g][j];
          b2t[j + 8] = (_Float16)acc[1][g][j];
        }
        v8f cz = {};
        v8f c2 = __builtin_amdgcn_wmma_f32_16x16x32_f16(false, xf[g], false, b2t,
                                                        (short)0, cz, false, false);
        float v = c2[0];
        v = (rsel == 1) ? c2[1] : v;
        v = (rsel == 2) ? c2[2] : v;
        v = (rsel == 3) ? c2[3] : v;
        v = (rsel == 4) ? c2[4] : v;
        v = (rsel == 5) ? c2[5] : v;
        v = (rsel == 6) ? c2[6] : v;
        v = (rsel == 7) ? c2[7] : v;
        if (diagActive && ok[g]) {
          v += bd[srcn[g] * F_OUT + o];                 // b3 contribution
          atomicAdd(&agg[dstn[g] * F_OUT + o], v);
        }
      }
    }
  }
}

// ---- finalize: relu(agg / max(deg,1)) ----
__launch_bounds__(256)
__global__ void ecc_final_kernel(const float* __restrict__ agg, const float* __restrict__ deg,
                                 float* __restrict__ out) {
  int i = blockIdx.x * blockDim.x + threadIdx.x;
  if (i < N_NODES * F_OUT) {
    float d = fmaxf(deg[i / F_OUT], 1.0f);
    out[i] = fmaxf(agg[i] / d, 0.0f);
  }
}

extern "C" void kernel_launch(void* const* d_in, const int* in_sizes, int n_in,
                              void* d_out, int out_size, void* d_ws, size_t ws_size,
                              hipStream_t stream) {
  const float* x         = (const float*)d_in[0];
  const float* edge_attr = (const float*)d_in[1];
  const float* W1 = (const float*)d_in[2];
  const float* b1 = (const float*)d_in[3];
  const float* W2 = (const float*)d_in[4];
  const float* b2 = (const float*)d_in[5];
  const float* W3 = (const float*)d_in[6];
  const float* b3 = (const float*)d_in[7];
  const int* esrc = (const int*)d_in[8];
  const int* edst = (const int*)d_in[9];

  char* ws = (char*)d_ws;
  _Float16* xh  = (_Float16*)(ws + XH_OFF);
  _Float16* h2g = (_Float16*)(ws + H2_OFF);
  float* agg = (float*)(ws + AGG_OFF);
  float* deg = (float*)(ws + DEG_OFF);
  float* bd  = (float*)(ws + BD_OFF);
  float* out = (float*)d_out;

  ecc_prep_kernel<<<(N_NODES * F_IN + 255) / 256, 256, 0, stream>>>(x, b3, xh, agg, deg, bd);
  ecc_deg_kernel<<<(N_EDGES + 255) / 256, 256, 0, stream>>>(edst, deg);
  ecc_mlp12_kernel<<<256, 256, 0, stream>>>(edge_attr, W1, b1, W2, b2, h2g);
  dim3 gC(224, 4, 1);
  ecc_main_kernel<<<gC, 256, 0, stream>>>(h2g, xh, W3, bd, esrc, edst, agg);
  ecc_final_kernel<<<(N_NODES * F_OUT + 255) / 256, 256, 0, stream>>>(agg, deg, out);
}
